// LTIinjection_30906584662348
// MI455X (gfx1250) — matrix operand order repaired and checked by reading, not compiled, not run on hardware
//
#include <hip/hip_runtime.h>
#include <hip/hip_bf16.h>

// ---------------------------------------------------------------------------
// LTI injection: u = x @ B^T  (bf16-split WMMA GEMM, TDM-staged, f32 accum)
//                h_t = a*h_{t-1} + u_t  (chunked parallel scan)
// Sizes: BATCH=8, T=2048, H=C=1024  ->  M=16384, N=1024, K=1024
//
// Pipeline:
//   0) split f32 x/B into bf16 hi/lo pairs (one streaming pass)
//   1) GEMM: TDM tensor_load_to_lds stages 128x32 bf16 tiles (with descriptor
//      padding reproducing the LDSK=40 layout), ds_load_b128 fragments,
//      v_wmma_f32_16x16x32_bf16 (hi*hi + hi*lo + lo*hi)
//   2-4) chunked parallel scan (carries -> cross-chunk states -> final scan)
// ---------------------------------------------------------------------------

#define N_EMBD 1024
#define BATCH  8
#define TLEN   2048
#define MTOT   (BATCH * TLEN)   // 16384

// GEMM tiling
#define BM 128
#define BN 128
#define BK 32
#define LDSK 40                 // padded stride (bf16 elems): 64B row + 16B pad

// scan chunking
#define NCHUNK 64
#define CLEN   32               // NCHUNK*CLEN == TLEN

typedef __bf16 bf16_t;
typedef __attribute__((ext_vector_type(16))) __bf16 v16bf;
typedef __attribute__((ext_vector_type(8)))  float  v8f;
typedef __attribute__((ext_vector_type(4)))  __bf16 v4bf;
typedef __attribute__((ext_vector_type(4)))  unsigned int v4u;
typedef __attribute__((ext_vector_type(8)))  unsigned int v8u;

union AFrag { v16bf v; uint4 q[2]; };

__device__ __forceinline__ float sigmoidf_dev(float v) {
    return 1.0f / (1.0f + __expf(-v));
}

// ---------------------------------------------------------------------------
// TDM 2D tile load: global (row-major, row length N_EMBD bf16) -> LDS,
// tile = BK elems x BM rows, with 16B pad after each 64B row => LDSK layout.
// Descriptor packing per CDNA5 ISA ch.8 (D# groups 0/1; groups 2/3 zero).
// ---------------------------------------------------------------------------
__device__ __forceinline__ void tdm_load_tile(void* ldsDst, const void* gsrc) {
    unsigned lds = (unsigned)reinterpret_cast<unsigned long long>(ldsDst); // AS(3) offset
    unsigned long long ga = reinterpret_cast<unsigned long long>(gsrc);
    v4u g0;
    g0[0] = 1u;                                   // count=1, user D#, no gather
    g0[1] = lds;                                  // lds_addr
    g0[2] = (unsigned)ga;                         // global_addr[31:0]
    g0[3] = (unsigned)((ga >> 32) & 0x1FFFFFFu)   // global_addr[56:32]
          | (2u << 30);                           // type = 2 ("image")
    v8u g1;
    g1[0] = (1u << 16)                            // data_size = 2 bytes
          | (1u << 20)                            // pad_enable
          | (3u << 22)                            // pad_interval: every 64B
          | (3u << 25);                           // pad_amount: 4 DWORDs = 16B
    g1[1] = (unsigned)(N_EMBD & 0xFFFF) << 16;    // tensor_dim0[15:0] @ bits63:48
    g1[2] = (unsigned)(N_EMBD >> 16)              // tensor_dim0[31:16]
          | (0x4000u << 16);                      // tensor_dim1[15:0] = 16384
    g1[3] = 0u | ((unsigned)BK << 16);            // tensor_dim1[31:16] | tile_dim0
    g1[4] = (unsigned)BM;                         // tile_dim1 (tile_dim2 = 0)
    g1[5] = (unsigned)N_EMBD;                     // tensor_dim0_stride[31:0]
    g1[6] = 0u;                                   // stride hi | dim1_stride lo
    g1[7] = 0u;                                   // dim1_stride hi
    v4u gz = {0u, 0u, 0u, 0u};                    // groups 2/3: dims unused
    asm volatile("tensor_load_to_lds %0, %1, %2, %3"
                 :
                 : "s"(g0), "s"(g1), "s"(gz), "s"(gz)
                 : "memory");
}

// ---------------------------------------------------------------------------
// Phase 0: split f32 -> bf16 hi + bf16 lo (lo = f - float(hi))
// ---------------------------------------------------------------------------
__global__ __launch_bounds__(256)
void lti_split_bf16(const float* __restrict__ src,
                    bf16_t* __restrict__ hiD,
                    bf16_t* __restrict__ loD, int n4) {
    int i = blockIdx.x * 256 + threadIdx.x;
    if (i >= n4) return;
    float4 f = reinterpret_cast<const float4*>(src)[i];
    v4bf hi, lo;
    hi.x = (__bf16)f.x; lo.x = (__bf16)(f.x - (float)hi.x);
    hi.y = (__bf16)f.y; lo.y = (__bf16)(f.y - (float)hi.y);
    hi.z = (__bf16)f.z; lo.z = (__bf16)(f.z - (float)hi.z);
    hi.w = (__bf16)f.w; lo.w = (__bf16)(f.w - (float)hi.w);
    reinterpret_cast<v4bf*>(hiD)[i] = hi;
    reinterpret_cast<v4bf*>(loD)[i] = lo;
}

// ---------------------------------------------------------------------------
// Phase 1: GEMM  U[m,h] = sum_c X[m,c]*B[h,c]
// TDM double-buffered staging; waves 0..3 each own one tile descriptor.
// ---------------------------------------------------------------------------
__global__ __launch_bounds__(256)
void lti_gemm_wmma(const bf16_t* __restrict__ Xh, const bf16_t* __restrict__ Xl,
                   const bf16_t* __restrict__ Bh, const bf16_t* __restrict__ Bl,
                   float* __restrict__ U) {
    __shared__ bf16_t Ash[2][BM * LDSK];
    __shared__ bf16_t Asl[2][BM * LDSK];
    __shared__ bf16_t Bsh[2][BN * LDSK];
    __shared__ bf16_t Bsl[2][BN * LDSK];

    const int tid   = threadIdx.x;
    const int lane  = tid & 31;
    const int wave  = tid >> 5;          // 0..7
    const int waveM = wave & 1;          // 2 waves along M (64 rows each)
    const int waveN = wave >> 1;         // 4 waves along N (32 cols each)

    const int m_block = blockIdx.y * BM;
    const int n_block = blockIdx.x * BN;

    const v8f vzero = {0.f, 0.f, 0.f, 0.f, 0.f, 0.f, 0.f, 0.f};
    v8f acc[4][2];
#pragma unroll
    for (int mi = 0; mi < 4; ++mi)
#pragma unroll
        for (int ni = 0; ni < 2; ++ni)
            acc[mi][ni] = vzero;

    const int l15   = lane & 15;
    const int lhalf = lane >> 4;               // 0 or 1
    const int kA    = lhalf << 3;              // A frag: K chunk 0 or 8 (elems)
    const int kB    = lhalf << 3;              // B frag: K chunk 0 or 8 (elems)

    // wave 0..3 -> (Xh->Ash), (Xl->Asl), (Bh->Bsh), (Bl->Bsl)
    const bf16_t* gsrc =
        (wave == 0) ? Xh : (wave == 1) ? Xl : (wave == 2) ? Bh : Bl;
    const size_t growbase =
        (size_t)((wave < 2) ? m_block : n_block) * N_EMBD;

    // prologue: DMA stage 0 into buffer 0
    if (wave < 4) {
        bf16_t* dst = (wave == 0) ? &Ash[0][0] : (wave == 1) ? &Asl[0][0]
                    : (wave == 2) ? &Bsh[0][0] : &Bsl[0][0];
        tdm_load_tile(dst, gsrc + growbase);
        __builtin_amdgcn_s_wait_tensorcnt(0);
    }
    __syncthreads();

    const int NK = N_EMBD / BK;                // 32 stages
    for (int kk = 0; kk < NK; ++kk) {
        const int buf = kk & 1;

        // issue next stage's DMA into the other buffer (overlaps WMMAs)
        if (kk + 1 < NK && wave < 4) {
            const int nb = buf ^ 1;
            bf16_t* dst = (wave == 0) ? &Ash[nb][0] : (wave == 1) ? &Asl[nb][0]
                        : (wave == 2) ? &Bsh[nb][0] : &Bsl[nb][0];
            tdm_load_tile(dst, gsrc + growbase + (size_t)(kk + 1) * BK);
        }

        // ---- build fragments (ISA 16-bit A 16x32 / B 32x16 layouts)
        v16bf ah[4], al[4], bh[2], bl[2];
#pragma unroll
        for (int mi = 0; mi < 4; ++mi) {
            int row = (waveM * 64 + mi * 16 + l15) * LDSK;
            AFrag fh, fl;
            fh.q[0] = *reinterpret_cast<const uint4*>(&Ash[buf][row + kA]);
            fh.q[1] = *reinterpret_cast<const uint4*>(&Ash[buf][row + kA + 16]);
            fl.q[0] = *reinterpret_cast<const uint4*>(&Asl[buf][row + kA]);
            fl.q[1] = *reinterpret_cast<const uint4*>(&Asl[buf][row + kA + 16]);
            ah[mi] = fh.v; al[mi] = fl.v;
        }
#pragma unroll
        for (int ni = 0; ni < 2; ++ni) {
            int row = (waveN * 32 + ni * 16 + l15) * LDSK;
            AFrag fh, fl;
            fh.q[0] = *reinterpret_cast<const uint4*>(&Bsh[buf][row + kB]);
            fh.q[1] = *reinterpret_cast<const uint4*>(&Bsh[buf][row + kB + 16]);
            fl.q[0] = *reinterpret_cast<const uint4*>(&Bsl[buf][row + kB]);
            fl.q[1] = *reinterpret_cast<const uint4*>(&Bsl[buf][row + kB + 16]);
            bh[ni] = fh.v; bl[ni] = fl.v;
        }

        // ---- 24 WMMAs per K-stage per wave (hi*hi + hi*lo + lo*hi)
#pragma unroll
        for (int mi = 0; mi < 4; ++mi) {
#pragma unroll
            for (int ni = 0; ni < 2; ++ni) {
                acc[mi][ni] = __builtin_amdgcn_wmma_f32_16x16x32_bf16(
                    false, ah[mi], false, bh[ni], (short)0, acc[mi][ni], false, false);
                acc[mi][ni] = __builtin_amdgcn_wmma_f32_16x16x32_bf16(
                    false, ah[mi], false, bl[ni], (short)0, acc[mi][ni], false, false);
                acc[mi][ni] = __builtin_amdgcn_wmma_f32_16x16x32_bf16(
                    false, al[mi], false, bh[ni], (short)0, acc[mi][ni], false, false);
            }
        }

        // issuing waves fence their DMA before the workgroup barrier
        if (kk + 1 < NK && wave < 4)
            __builtin_amdgcn_s_wait_tensorcnt(0);
        __syncthreads();
    }

    // ---- store C/D tiles (VGPR r: lanes0-15 -> M=r, lanes16-31 -> M=r+8)
#pragma unroll
    for (int mi = 0; mi < 4; ++mi) {
#pragma unroll
        for (int ni = 0; ni < 2; ++ni) {
            int mrow = m_block + waveM * 64 + mi * 16 + (lhalf << 3);
            int ncol = n_block + waveN * 32 + ni * 16 + l15;
            float* dst = &U[(size_t)mrow * N_EMBD + ncol];
#pragma unroll
            for (int r = 0; r < 8; ++r)
                dst[(size_t)r * N_EMBD] = acc[mi][ni][r];
        }
    }
}

// ---------------------------------------------------------------------------
// Scan phase 1: per-chunk carries  carry = sum_i a^(CLEN-1-i) * u_i
// ---------------------------------------------------------------------------
__global__ __launch_bounds__(256)
void lti_scan_carry(const float* __restrict__ U,
                    const float* __restrict__ raw_a,
                    float* __restrict__ carry) {
    int id    = blockIdx.x * 256 + threadIdx.x;    // b*65536 + chunk*1024 + h
    int h     = id & (N_EMBD - 1);
    int chunk = (id >> 10) & (NCHUNK - 1);
    int b     = id >> 16;
    float a = sigmoidf_dev(raw_a[h]);
    const float* up = U + (size_t)(b * TLEN + chunk * CLEN) * N_EMBD + h;
    float s = 0.0f;
#pragma unroll
    for (int i = 0; i < CLEN; ++i) {
        if (i + 12 < CLEN)
            __builtin_prefetch(up + (size_t)(i + 12) * N_EMBD, 0, 0);
        s = fmaf(a, s, up[(size_t)i * N_EMBD]);
    }
    carry[id] = s;
}

// ---------------------------------------------------------------------------
// Scan phase 2: cross-chunk states  s_{j+1} = a^CLEN * s_j + carry_j
// ---------------------------------------------------------------------------
__global__ __launch_bounds__(256)
void lti_scan_state(const float* __restrict__ carry,
                    const float* __restrict__ raw_a,
                    const float* __restrict__ h0,
                    float* __restrict__ S) {
    int id = blockIdx.x * 256 + threadIdx.x;       // b*1024 + h
    int h  = id & (N_EMBD - 1);
    int b  = id >> 10;
    float a  = sigmoidf_dev(raw_a[h]);
    float a2 = a * a, a4 = a2 * a2, a8 = a4 * a4, a16 = a8 * a8;
    float aL = a16 * a16;                          // a^32
    float s = h0[id];
    for (int j = 0; j < NCHUNK; ++j) {
        size_t idx = (size_t)(b * NCHUNK + j) * N_EMBD + h;
        S[idx] = s;
        s = fmaf(aL, s, carry[idx]);
    }
}

// ---------------------------------------------------------------------------
// Scan phase 3: final per-chunk scan, seeded with entering state; writes y
// ---------------------------------------------------------------------------
__global__ __launch_bounds__(256)
void lti_scan_final(const float* __restrict__ U,
                    const float* __restrict__ raw_a,
                    const float* __restrict__ S,
                    float* __restrict__ Y) {
    int id    = blockIdx.x * 256 + threadIdx.x;
    int h     = id & (N_EMBD - 1);
    int chunk = (id >> 10) & (NCHUNK - 1);
    int b     = id >> 16;
    float a = sigmoidf_dev(raw_a[h]);
    float s = S[id];
    const float* up = U + (size_t)(b * TLEN + chunk * CLEN) * N_EMBD + h;
    float*       yp = Y + (size_t)(b * TLEN + chunk * CLEN) * N_EMBD + h;
#pragma unroll
    for (int i = 0; i < CLEN; ++i) {
        if (i + 12 < CLEN)
            __builtin_prefetch(up + (size_t)(i + 12) * N_EMBD, 0, 0);
        s = fmaf(a, s, up[(size_t)i * N_EMBD]);
        yp[(size_t)i * N_EMBD] = s;
    }
}

// ---------------------------------------------------------------------------
extern "C" void kernel_launch(void* const* d_in, const int* in_sizes, int n_in,
                              void* d_out, int out_size, void* d_ws, size_t ws_size,
                              hipStream_t stream) {
    const float* x     = (const float*)d_in[0];   // [8, 2048, 1024]
    const float* raw_a = (const float*)d_in[1];   // [1024]
    const float* Bm    = (const float*)d_in[2];   // [1024, 1024]
    const float* h0    = (const float*)d_in[3];   // [8, 1024]
    float* Y = (float*)d_out;                     // [8, 2048, 1024]

    // workspace layout (needs ~136 MB)
    char* w = (char*)d_ws;
    float*  U     = (float*)w;                                    // 64 MB
    bf16_t* Xh    = (bf16_t*)(w + ((size_t)MTOT * N_EMBD * 4));   // 32 MB
    bf16_t* Xl    = Xh + (size_t)MTOT * N_EMBD;                   // 32 MB
    bf16_t* Bh    = Xl + (size_t)MTOT * N_EMBD;                   // 2 MB
    bf16_t* Bl    = Bh + (size_t)N_EMBD * N_EMBD;                 // 2 MB
    float*  carry = (float*)(Bl + (size_t)N_EMBD * N_EMBD);       // 2 MB
    float*  S     = carry + (size_t)BATCH * NCHUNK * N_EMBD;      // 2 MB

    // phase 0: split f32 -> bf16 hi/lo
    int nx4 = (MTOT * N_EMBD) / 4;                // 4194304
    int nb4 = (N_EMBD * N_EMBD) / 4;              // 262144
    lti_split_bf16<<<nx4 / 256, 256, 0, stream>>>(x, Xh, Xl, nx4);
    lti_split_bf16<<<nb4 / 256, 256, 0, stream>>>(Bm, Bh, Bl, nb4);

    // phase 1: WMMA GEMM with TDM staging
    dim3 ggrid(N_EMBD / BN, MTOT / BM);           // (8, 128)
    lti_gemm_wmma<<<ggrid, 256, 0, stream>>>(Xh, Xl, Bh, Bl, U);

    // phases 2-4: chunked scan
    int nchan = BATCH * NCHUNK * N_EMBD;          // 524288
    lti_scan_carry<<<nchan / 256, 256, 0, stream>>>(U, raw_a, carry);
    lti_scan_state<<<(BATCH * N_EMBD) / 256, 256, 0, stream>>>(carry, raw_a, h0, S);
    lti_scan_final<<<nchan / 256, 256, 0, stream>>>(U, raw_a, S, Y);
}
